// PagedAttention_28638841930372
// MI455X (gfx1250) — matrix operand order repaired and checked by reading, not compiled, not run on hardware
//
#include <hip/hip_runtime.h>
#include <hip/hip_bf16.h>

// ---------------- problem constants (match reference) ----------------
#define B_SZ   32
#define H_Q    32
#define H_KV   8
#define GQ     4            // H_Q / H_KV
#define DHEAD  128
#define BS     16           // paged-cache block size
#define BLK_PER_SEQ 128
#define MAX_CTX (BLK_PER_SEQ*BS)    // 2048
// flash-decoding configuration: 32-token (2-page) tiles per iteration
#define TOK    32
#define SPLITS 4
#define NWAVES 2
#define ENTRIES (SPLITS*NWAVES)          // 8 partials per (b,kv)
#define CH32_PER_SPLIT (MAX_CTX/TOK/SPLITS)  // 16 double-pages per split

typedef __bf16 v16bf __attribute__((ext_vector_type(16)));
typedef float  v8f   __attribute__((ext_vector_type(8)));
typedef unsigned int u32x4 __attribute__((ext_vector_type(4)));
typedef int i32x8 __attribute__((ext_vector_type(8)));
typedef int i32x4 __attribute__((ext_vector_type(4)));

__device__ __forceinline__ float bf16r(float x) { return (float)(__bf16)x; }

// 16-lane butterfly reductions via DPP (no LDS traffic, no dscnt waits).
// quad_perm xor1 (0xB1), quad_perm xor2 (0x4E), row_half_mirror (0x141),
// row_mirror (0x140) -> full reduction within each 16-lane row.
__device__ __forceinline__ float red16_max(float x) {
    x = fmaxf(x, __int_as_float(__builtin_amdgcn_update_dpp(0, __float_as_int(x), 0xB1,  0xF, 0xF, true)));
    x = fmaxf(x, __int_as_float(__builtin_amdgcn_update_dpp(0, __float_as_int(x), 0x4E,  0xF, 0xF, true)));
    x = fmaxf(x, __int_as_float(__builtin_amdgcn_update_dpp(0, __float_as_int(x), 0x141, 0xF, 0xF, true)));
    x = fmaxf(x, __int_as_float(__builtin_amdgcn_update_dpp(0, __float_as_int(x), 0x140, 0xF, 0xF, true)));
    return x;
}
__device__ __forceinline__ float red16_sum(float x) {
    x += __int_as_float(__builtin_amdgcn_update_dpp(0, __float_as_int(x), 0xB1,  0xF, 0xF, true));
    x += __int_as_float(__builtin_amdgcn_update_dpp(0, __float_as_int(x), 0x4E,  0xF, 0xF, true));
    x += __int_as_float(__builtin_amdgcn_update_dpp(0, __float_as_int(x), 0x141, 0xF, 0xF, true));
    x += __int_as_float(__builtin_amdgcn_update_dpp(0, __float_as_int(x), 0x140, 0xF, 0xF, true));
    return x;
}

// LDS per wave: K tile (2 pages fp32) | V^T tile (bf16) | P buffer (bf16)
#define K_BYTES  (TOK*DHEAD*4)    // 16384
#define VT_BYTES (DHEAD*TOK*2)    // 8192
#define P_BYTES  (16*TOK*2)       // 1024
#define PW       (K_BYTES + VT_BYTES + P_BYTES)   // 25600 * 2 waves = 51200 B

// TDM: async DMA of one K page (16 tokens x 128 fp32, row stride 1024 elems)
__device__ __forceinline__ void tdm_load_page(const float* base, int blk, int kv,
                                              unsigned lds_off)
{
    unsigned long long ga = (unsigned long long)(const void*)base
        + ((((unsigned long long)blk * BS) * H_KV + kv) * DHEAD) * 4ull;
    u32x4 g0; i32x8 g1; i32x4 g2, g3; i32x8 g4;
    g0[0] = 1u;                                        // count=1 (user D#)
    g0[1] = lds_off;                                   // lds_addr
    g0[2] = (unsigned)(ga & 0xffffffffu);              // global_addr[31:0]
    g0[3] = (unsigned)((ga >> 32) & 0x01ffffffu) | (2u << 30);  // [56:32] | type=2
    g1[0] = (2 << 16);                                 // data_size = 4B
    g1[1] = (DHEAD << 16);                             // tensor_dim0 = 128
    g1[2] = (BS << 16);                                // tensor_dim1 = 16
    g1[3] = (DHEAD << 16);                             // tile_dim0 = 128
    g1[4] = BS;                                        // tile_dim1 = 16
    g1[5] = H_KV * DHEAD;                              // dim0_stride = 1024 elems
    g1[6] = 0; g1[7] = 0;
    g2[0] = g2[1] = g2[2] = g2[3] = 0;
    g3[0] = g3[1] = g3[2] = g3[3] = 0;
    g4[0] = g4[1] = g4[2] = g4[3] = 0;
    g4[4] = g4[5] = g4[6] = g4[7] = 0;
    __builtin_amdgcn_tensor_load_to_lds(g0, g1, g2, g3, g4, 0);
}

// ---------------- kernel 1: RoPE for q and the new k ----------------
__global__ __launch_bounds__(DHEAD) void rope_prep(
    const float* __restrict__ q, const float* __restrict__ k_new,
    const int* __restrict__ ctx, float* __restrict__ q_rot,
    float* __restrict__ k_rot)
{
    const int b = blockIdx.x, d = threadIdx.x;
    const int j = d & 63;  // concat([freqs, freqs]) duplicates the 64 freqs
    const float inv = bf16r(__powf(10000.0f, -(float)j * (1.0f / 64.0f)));
    const float t   = bf16r((float)ctx[b]);
    const float fr  = bf16r(t * inv);
    const float cs  = bf16r(cosf(fr));
    const float sn  = bf16r(sinf(fr));
    #pragma unroll
    for (int h = 0; h < H_Q; ++h) {
        const float* x = q + (size_t)(b * H_Q + h) * DHEAD;
        const float rot = (d < 64) ? -x[d + 64] : x[d - 64];
        q_rot[(size_t)(b * H_Q + h) * DHEAD + d] = x[d] * cs + rot * sn;
    }
    #pragma unroll
    for (int h = 0; h < H_KV; ++h) {
        const float* x = k_new + (size_t)(b * H_KV + h) * DHEAD;
        const float rot = (d < 64) ? -x[d + 64] : x[d - 64];
        k_rot[(size_t)(b * H_KV + h) * DHEAD + d] = x[d] * cs + rot * sn;
    }
}

// ---------------- kernel 2: flash-decoding paged attention ----------------
__global__ __launch_bounds__(NWAVES*32) void paged_attn_fd(
    const float* __restrict__ q_rot, const float* __restrict__ k_rot,
    const float* __restrict__ v_new,
    const float* __restrict__ k_cache, const float* __restrict__ v_cache,
    const int* __restrict__ block_table, const int* __restrict__ ctx_lens,
    float* __restrict__ pO, float* __restrict__ pM, float* __restrict__ pL)
{
    __shared__ __align__(16) unsigned char smem[NWAVES * PW];

    const int split = blockIdx.x, kv = blockIdx.y, b = blockIdx.z;
    const int wave = threadIdx.x >> 5, lane = threadIdx.x & 31;
    const int kh = lane >> 4;        // lane-half: K-range selector for WMMA operands
    const int l15 = lane & 15;
    const int pos = ctx_lens[b];
    const int nch = (pos + TOK - 1) >> 5;          // 32-token tiles in context
    const int c_lo = split * CH32_PER_SPLIT;
    const int c_hi_full = c_lo + CH32_PER_SPLIT;
    const int c_hi = nch < c_hi_full ? nch : c_hi_full;

    float*  Kw = (float*)(smem + wave * PW);                       // [32][128] f32
    __bf16* Vt = (__bf16*)(smem + wave * PW + K_BYTES);            // [128][32] bf16
    __bf16* PB = (__bf16*)(smem + wave * PW + K_BYTES + VT_BYTES); // [16][32]  bf16
    const unsigned ldsK_off = (unsigned)(wave * PW);  // single __shared__ -> LDS offset 0

    // zero the P staging buffer once (padding rows 4..15 must stay zero)
    {
        const uint4 z = {0u, 0u, 0u, 0u};
        reinterpret_cast<uint4*>(PB)[lane]      = z;
        reinterpret_cast<uint4*>(PB)[lane + 32] = z;
    }

    // ---- Q as 4 A-operands (K-dim chunks of 32 across D=128), bf16 ----
    v16bf Aq[4];
    {
        const bool mvalid = (l15 < GQ);       // rows 4..15 are zero padding
        const int hq = kv * GQ + (l15 < GQ ? l15 : 0);
        const float* qr = q_rot + (size_t)(b * H_Q + hq) * DHEAD;
        #pragma unroll
        for (int c4 = 0; c4 < 4; ++c4) {
            #pragma unroll
            for (int i = 0; i < 16; ++i) {
                const int k = c4 * 32 + kh * 16 + i;
                Aq[c4][i] = mvalid ? (__bf16)qr[k] : (__bf16)0.0f;
            }
        }
    }

    float mrun[GQ], lrun[GQ];
    v8f O[8];
    #pragma unroll
    for (int r = 0; r < GQ; ++r) { mrun[r] = -1e30f; lrun[r] = 0.0f; }
    #pragma unroll
    for (int nt = 0; nt < 8; ++nt)
        #pragma unroll
        for (int r = 0; r < 8; ++r) O[nt][r] = 0.0f;

    const float scale = 0.08838834764831845f;   // 128^-0.5
    __builtin_amdgcn_wave_barrier();

    for (int c = c_lo + wave; c < c_hi; c += NWAVES) {
        const int blk0 = __builtin_amdgcn_readfirstlane(block_table[b * BLK_PER_SEQ + 2*c]);
        const int blk1 = __builtin_amdgcn_readfirstlane(block_table[b * BLK_PER_SEQ + 2*c + 1]);

        // async DMA both K pages of this 32-token tile into LDS
        tdm_load_page(k_cache, blk0, kv, ldsK_off);
        tdm_load_page(k_cache, blk1, kv, ldsK_off + K_BYTES/2);

        // prefetch next iteration's K/V pages toward L2 (global_prefetch_b8)
        if (c + NWAVES < c_hi) {
            const int nb0 = block_table[b * BLK_PER_SEQ + 2*(c + NWAVES)];
            const int nb1 = block_table[b * BLK_PER_SEQ + 2*(c + NWAVES) + 1];
            const size_t roff = (size_t)(lane >> 1) * (H_KV*DHEAD) + (size_t)(lane & 1) * 64;
            __builtin_prefetch(k_cache + ((size_t)nb0*BS*H_KV + kv)*DHEAD + roff, 0, 1);
            __builtin_prefetch(v_cache + ((size_t)nb0*BS*H_KV + kv)*DHEAD + roff, 0, 1);
            __builtin_prefetch(k_cache + ((size_t)nb1*BS*H_KV + kv)*DHEAD + roff, 0, 1);
            __builtin_prefetch(v_cache + ((size_t)nb1*BS*H_KV + kv)*DHEAD + roff, 0, 1);
        }

        // ---- V pages: global->reg, bf16 transpose into LDS (overlaps the TDM) ----
        #pragma unroll
        for (int hp = 0; hp < 2; ++hp) {
            const int blk = hp ? blk1 : blk0;
            const float* vbase = v_cache + (((size_t)blk * BS) * H_KV + kv) * DHEAD;
            #pragma unroll
            for (int t = 0; t < BS; ++t) {
                const float4 f = reinterpret_cast<const float4*>(vbase + (size_t)t*H_KV*DHEAD)[lane];
                __bf16* dst = Vt + hp*BS + t;            // Vt[d][tok], row stride TOK
                dst[(lane*4 + 0) * TOK] = (__bf16)f.x;
                dst[(lane*4 + 1) * TOK] = (__bf16)f.y;
                dst[(lane*4 + 2) * TOK] = (__bf16)f.z;
                dst[(lane*4 + 3) * TOK] = (__bf16)f.w;
            }
        }
        // substitute new token's V (cache is read-only; scatter emulated here)
        if ((unsigned)(pos - 1 - c*TOK) < (unsigned)TOK) {
            const int t = pos - 1 - c*TOK;
            const float4 f = reinterpret_cast<const float4*>(v_new + (size_t)(b*H_KV + kv)*DHEAD)[lane];
            Vt[(lane*4 + 0)*TOK + t] = (__bf16)f.x;
            Vt[(lane*4 + 1)*TOK + t] = (__bf16)f.y;
            Vt[(lane*4 + 2)*TOK + t] = (__bf16)f.z;
            Vt[(lane*4 + 3)*TOK + t] = (__bf16)f.w;
        }

        __builtin_amdgcn_s_wait_tensorcnt(0);
        // substitute the freshly RoPE'd key for token pos-1
        if ((unsigned)(pos - 1 - c*TOK) < (unsigned)TOK) {
            const int t = pos - 1 - c*TOK;
            reinterpret_cast<float4*>(Kw + t*DHEAD)[lane] =
                reinterpret_cast<const float4*>(k_rot + (size_t)(b*H_KV + kv)*DHEAD)[lane];
        }
        __builtin_amdgcn_wave_barrier();   // order intra-wave LDS writes vs reads

        // ---- scores: two 16x16 tiles, 4x v_wmma_f32_16x16x32_bf16 each ----
        v8f S0 = {0.f,0.f,0.f,0.f,0.f,0.f,0.f,0.f};
        v8f S1 = {0.f,0.f,0.f,0.f,0.f,0.f,0.f,0.f};
        #pragma unroll
        for (int c4 = 0; c4 < 4; ++c4) {
            const float* kp0 = Kw + l15*DHEAD + c4*32 + kh*16;
            v16bf Bk;
            #pragma unroll
            for (int i = 0; i < 16; ++i) Bk[i] = (__bf16)kp0[i];
            S0 = __builtin_amdgcn_wmma_f32_16x16x32_bf16(false, Aq[c4], false, Bk,
                                                         (short)0, S0, false, false);
            const float* kp1 = kp0 + BS*DHEAD;
            #pragma unroll
            for (int i = 0; i < 16; ++i) Bk[i] = (__bf16)kp1[i];
            S1 = __builtin_amdgcn_wmma_f32_16x16x32_bf16(false, Aq[c4], false, Bk,
                                                         (short)0, S1, false, false);
        }

        // ---- masked online softmax (real rows 0..3 only; DPP reductions) ----
        const bool tv0 = (c*TOK + l15) < pos;
        const bool tv1 = (c*TOK + 16 + l15) < pos;
        float alpha[GQ];
        #pragma unroll
        for (int r = 0; r < GQ; ++r) {
            const float s0 = tv0 ? S0[r]*scale : -1e30f;
            const float s1 = tv1 ? S1[r]*scale : -1e30f;
            const float rm = red16_max(fmaxf(s0, s1));
            const float mn = fmaxf(mrun[r], rm);
            const float al = __expf(mrun[r] - mn);
            const float p0 = __expf(s0 - mn);
            const float p1 = __expf(s1 - mn);
            const float rs = red16_sum(p0 + p1);
            lrun[r] = lrun[r]*al + rs;
            mrun[r] = mn;
            alpha[r] = al;
            if (kh == 0) {
                PB[r*TOK + l15]      = (__bf16)p0;
                PB[r*TOK + 16 + l15] = (__bf16)p1;
            }
        }
        __builtin_amdgcn_wave_barrier();

        #pragma unroll
        for (int nt = 0; nt < 8; ++nt)
            #pragma unroll
            for (int r = 0; r < GQ; ++r) O[nt][r] *= alpha[r];

        // ---- O += P V : A = P (16x32, full K), B = V^T tiles (full K) ----
        v16bf Ap;
        {
            const __bf16* pr = PB + l15*TOK;
            #pragma unroll
            for (int i = 0; i < 16; ++i) Ap[i] = pr[kh*16 + i];
        }
        #pragma unroll
        for (int nt = 0; nt < 8; ++nt) {
            v16bf Bv;
            const __bf16* vr = Vt + (nt*16 + l15)*TOK + kh*16;
            #pragma unroll
            for (int i = 0; i < 16; ++i) Bv[i] = vr[i];
            O[nt] = __builtin_amdgcn_wmma_f32_16x16x32_bf16(false, Ap, false, Bv,
                                                            (short)0, O[nt], false, false);
        }
        __builtin_amdgcn_wave_barrier();   // before next iter overwrites LDS
    }

    // ---- spill flash-decoding partials ----
    const int e = split * NWAVES + wave;
    const size_t rbase = (((size_t)e * B_SZ + b) * H_KV + kv) * GQ;
    if (kh == 0) {
        #pragma unroll
        for (int r = 0; r < GQ; ++r)
            #pragma unroll
            for (int nt = 0; nt < 8; ++nt)
                pO[(rbase + r) * DHEAD + nt*16 + l15] = O[nt][r];
    }
    if (lane == 0) {
        #pragma unroll
        for (int r = 0; r < GQ; ++r) { pM[rbase + r] = mrun[r]; pL[rbase + r] = lrun[r]; }
    }
}

// ---------------- kernel 3: combine partials ----------------
__global__ __launch_bounds__(DHEAD) void fd_combine(
    const float* __restrict__ pO, const float* __restrict__ pM,
    const float* __restrict__ pL, float* __restrict__ out)
{
    const int row = blockIdx.x;     // b*H + h  (h = kv*GQ + g)
    const int d = threadIdx.x;
    const int NR = B_SZ * H_Q;
    float M = -1e30f;
    #pragma unroll
    for (int e = 0; e < ENTRIES; ++e) M = fmaxf(M, pM[(size_t)e * NR + row]);
    float L = 0.0f, acc = 0.0f;
    #pragma unroll
    for (int e = 0; e < ENTRIES; ++e) {
        const float w = __expf(pM[(size_t)e * NR + row] - M);
        L   += w * pL[(size_t)e * NR + row];
        acc += w * pO[((size_t)e * NR + row) * DHEAD + d];
    }
    out[(size_t)row * DHEAD + d] = acc / L;
}

// ---------------- launcher ----------------
extern "C" void kernel_launch(void* const* d_in, const int* in_sizes, int n_in,
                              void* d_out, int out_size, void* d_ws, size_t ws_size,
                              hipStream_t stream)
{
    (void)in_sizes; (void)n_in; (void)out_size; (void)ws_size;
    const float* q       = (const float*)d_in[0];
    const float* k_new   = (const float*)d_in[1];
    const float* v_new   = (const float*)d_in[2];
    const float* k_cache = (const float*)d_in[3];
    const float* v_cache = (const float*)d_in[4];
    const int*   btab    = (const int*)d_in[5];
    const int*   ctx     = (const int*)d_in[6];
    float* out = (float*)d_out;

    float* q_rot = (float*)d_ws;                                   // 131072 f
    float* k_rot = q_rot + (size_t)B_SZ * H_Q * DHEAD;             // 32768 f
    float* pO    = k_rot + (size_t)B_SZ * H_KV * DHEAD;            // 8*131072 f
    float* pM    = pO + (size_t)ENTRIES * B_SZ * H_Q * DHEAD;      // 8192 f
    float* pL    = pM + (size_t)ENTRIES * B_SZ * H_Q;              // 8192 f

    rope_prep<<<B_SZ, DHEAD, 0, stream>>>(q, k_new, ctx, q_rot, k_rot);
    paged_attn_fd<<<dim3(SPLITS, H_KV, B_SZ), NWAVES * 32, 0, stream>>>(
        q_rot, k_rot, v_new, k_cache, v_cache, btab, ctx, pO, pM, pL);
    fd_combine<<<B_SZ * H_Q, DHEAD, 0, stream>>>(pO, pM, pL, out);
}
